// GATNet_63350767616010
// MI455X (gfx1250) — compile-verified, matrix operand order checked by reading, not run on hardware
//
#include <hip/hip_runtime.h>
#include <hip/hip_bf16.h>

// ---------------------------------------------------------------------------
// GATNet on MI455X (gfx1250): bf16 WMMA GEMMs with async global->LDS A-tile
// staging (ASYNCcnt) + one-shot LDS-side bf16 conversion, + atomic scatter
// GAT layers.
// ---------------------------------------------------------------------------

#define HID 128
#define HEADS 4
#define CH 32
#define TM 32      // rows per block in GEMM
#define KC 64      // K-chunk staged in LDS

typedef __attribute__((ext_vector_type(16))) __bf16 v16bf;
typedef __attribute__((ext_vector_type(8)))  float  v8f;

union V16 {
    v16bf v;
    uint4 q[2];
};

// float -> bf16, round-to-nearest-even
__device__ __forceinline__ unsigned short f2bf(float f) {
    unsigned u = __float_as_uint(f);
    unsigned r = u + 0x7FFFu + ((u >> 16) & 1u);
    return (unsigned short)(r >> 16);
}
// pack two floats into one dword of bf16 pairs
__device__ __forceinline__ unsigned pk2(float a, float b) {
    return (unsigned)f2bf(a) | ((unsigned)f2bf(b) << 16);
}

__device__ __forceinline__ float gelu_exact(float v) {
    return 0.5f * v * (1.0f + erff(v * 0.70710678118654752f));
}

// order-preserving float <-> uint key (for atomicMax-based segment max)
__device__ __forceinline__ unsigned f2key(float f) {
    unsigned u = __float_as_uint(f);
    return (u & 0x80000000u) ? ~u : (u | 0x80000000u);
}
__device__ __forceinline__ float key2f(unsigned k) {
    unsigned u = (k & 0x80000000u) ? (k & 0x7FFFFFFFu) : ~k;
    return __uint_as_float(u);
}
#define KEY_NEG_INF 0x007FFFFFu   // f2key(-inf)

// low 32 bits of a flat pointer to __shared__ == LDS byte address
__device__ __forceinline__ unsigned lds_addr(const void* p) {
    return (unsigned)(uintptr_t)p;
}

// ---------------------------------------------------------------------------
// GEMM: C[nrows,128] = epi(A[nrows,Kdim] @ B[Kdim,128] (+ bias))
// EPI==1: bias + exact GELU.  EPI==0: raw output.
//
// Per K-chunk:
//   1) each thread issues global_load_async_to_lds_b128 for its A sub-tile
//   2) B tile converted to bf16, stored K-transposed in LDS (overlaps async)
//   3) s_wait_asynccnt 0 -> each wave converts the f32 region its own lanes
//      loaded into bf16 (no barrier needed before this step)
//   4) one barrier, then a conversion-free inner loop:
//      ds_load_b128 fragments -> v_wmma_f32_16x16x32_bf16
//
// 8 waves/block; wave w: M-half = w>>2, 32-col group = w&3 (two 16x16 tiles).
// ---------------------------------------------------------------------------
template <int EPI>
__global__ __launch_bounds__(256)
void gemm_wmma_bf16(const float* __restrict__ A, const float* __restrict__ B,
                    const float* __restrict__ bias, float* __restrict__ C,
                    int nrows, int Kdim) {
    __shared__ __align__(16) float          Af32[TM][KC];   // raw f32 A tile (8KB)
    __shared__ __align__(16) unsigned short Abf[TM][KC];    // bf16 A tile (4KB)
    __shared__ __align__(16) unsigned short Bsb[HID][KC];   // [n][k] bf16 (16KB)

    const int t    = threadIdx.x;
    const int wave = t >> 5;
    const int lane = t & 31;
    const int mt   = wave >> 2;      // 0..1
    const int ng   = wave & 3;       // 0..3
    const int rowBase = blockIdx.x * TM;

    const int r  = t >> 4;           // 0..15  (A staging row within 16-row pass)
    const int c4 = (t & 15) << 2;    // 0,4,...,60

    v8f acc0 = {};
    v8f acc1 = {};

    const int nChunks = Kdim / KC;
    for (int chn = 0; chn < nChunks; ++chn) {
        const int k0 = chn * KC;

        // ---- 1) async-stage A tile: TM x KC f32 straight into LDS -------
        // 8KB / 256 threads = 2 x b128 per thread.
        #pragma unroll
        for (int p = 0; p < 2; ++p) {
            const int row = p * 16 + r;
            const int gr  = rowBase + row;
            if (gr < nrows) {
                const unsigned     dst = lds_addr(&Af32[row][c4]);
                const float* const src = A + (size_t)gr * Kdim + k0 + c4;
                asm volatile("global_load_async_to_lds_b128 %0, %1, off"
                             :: "v"(dst), "v"(src) : "memory");
            }
        }
        // ---- 2) stage B tile: KC x 128 f32 -> bf16, transposed [n][k] ----
        #pragma unroll
        for (int p = 0; p < 8; ++p) {
            const int lin = p * 256 + t;        // 0..2047
            const int kk  = lin >> 5;           // 0..63
            const int cb  = (lin & 31) << 2;    // 0,4,...,124
            const float4 v = *(const float4*)(B + (size_t)(k0 + kk) * HID + cb);
            Bsb[cb + 0][kk] = f2bf(v.x);
            Bsb[cb + 1][kk] = f2bf(v.y);
            Bsb[cb + 2][kk] = f2bf(v.z);
            Bsb[cb + 3][kk] = f2bf(v.w);
        }
        // ---- 3) wait own async loads, convert own region f32 -> bf16 ----
        asm volatile("s_wait_asynccnt 0x0" ::: "memory");
        #pragma unroll
        for (int p = 0; p < 2; ++p) {
            const int row = p * 16 + r;
            const float4 f0 = *(const float4*)&Af32[row][c4];
            const float4 f1 = *(const float4*)&Af32[row][c4 + 4];
            *(uint2*)&Abf[row][c4]     = make_uint2(pk2(f0.x, f0.y), pk2(f0.z, f0.w));
            *(uint2*)&Abf[row][c4 + 4] = make_uint2(pk2(f1.x, f1.y), pk2(f1.z, f1.w));
        }
        __syncthreads();

        // ---- 4) two 16x16x32 WMMA steps per chunk (conversion-free) -----
        const int hi8  = (lane >> 4) << 3;        // 0 or 8 (K sub-offset)
        const int mrow = mt * 16 + (lane & 15);
        const int n0   = ng * 32 + (lane & 15);
        #pragma unroll
        for (int ks = 0; ks < KC; ks += 32) {
            V16 a, b0, b1;
            // A frag: lanes 0-15: K = ks+[0..7], ks+16+[0..7]; lanes 16-31: +8
            a.q[0]  = *(const uint4*)&Abf[mrow][ks + hi8];
            a.q[1]  = *(const uint4*)&Abf[mrow][ks + 16 + hi8];
            // B frags: same K pattern, N = lane&15 (+16 for second tile)
            b0.q[0] = *(const uint4*)&Bsb[n0][ks + hi8];
            b0.q[1] = *(const uint4*)&Bsb[n0][ks + 16 + hi8];
            b1.q[0] = *(const uint4*)&Bsb[n0 + 16][ks + hi8];
            b1.q[1] = *(const uint4*)&Bsb[n0 + 16][ks + 16 + hi8];
            acc0 = __builtin_amdgcn_wmma_f32_16x16x32_bf16(
                false, a.v, false, b0.v, (short)0, acc0, false, false);
            acc1 = __builtin_amdgcn_wmma_f32_16x16x32_bf16(
                false, a.v, false, b1.v, (short)0, acc1, false, false);
        }
        __syncthreads();
    }

    // ---- epilogue: C/D layout lanes 16-31 -> M = 8 + r ----
    const int mBase = rowBase + mt * 16 + ((lane >> 4) << 3);
    const int n0    = ng * 32 + (lane & 15);
    #pragma unroll
    for (int rr = 0; rr < 8; ++rr) {
        const int m = mBase + rr;
        if (m >= nrows) break;
        float v0 = acc0[rr];
        float v1 = acc1[rr];
        if (EPI == 1) {
            v0 = gelu_exact(v0 + bias[n0]);
            v1 = gelu_exact(v1 + bias[n0 + 16]);
        }
        C[(size_t)m * HID + n0]      = v0;
        C[(size_t)m * HID + n0 + 16] = v1;
    }
}

// ---------------------------------------------------------------------------
// alpha_s[n,h] = <h[n,h,:], a_src[h,:]>, alpha_d likewise
// ---------------------------------------------------------------------------
__global__ void alpha_kernel(const float* __restrict__ h,
                             const float* __restrict__ asrc,
                             const float* __restrict__ adst,
                             float* __restrict__ als, float* __restrict__ ald,
                             int Nn) {
    const int tid = blockIdx.x * blockDim.x + threadIdx.x;
    if (tid >= Nn * HEADS) return;
    const int n  = tid >> 2;
    const int hd = tid & 3;
    const float* hp = h + (size_t)n * HID + hd * CH;
    const float* as = asrc + hd * CH;
    const float* ad = adst + hd * CH;
    float ss = 0.f, sd = 0.f;
    #pragma unroll
    for (int c = 0; c < CH; ++c) {
        const float v = hp[c];
        ss += v * as[c];
        sd += v * ad[c];
    }
    als[tid] = ss;
    ald[tid] = sd;
}

// ---------------------------------------------------------------------------
// fills
// ---------------------------------------------------------------------------
__global__ void fill_f32_kernel(float* p, float v, int n) {
    const int i = blockIdx.x * blockDim.x + threadIdx.x;
    if (i < n) p[i] = v;
}
__global__ void fill_u32_kernel(unsigned* p, unsigned v, int n) {
    const int i = blockIdx.x * blockDim.x + threadIdx.x;
    if (i < n) p[i] = v;
}

// ---------------------------------------------------------------------------
// edge pass 1: e = leakyrelu(alpha_s[src]+alpha_d[dst]); segment max via
// order-preserving uint key + atomicMax.
// ---------------------------------------------------------------------------
__global__ void edge_max_kernel(const long long* __restrict__ ei, int E, int Etot,
                                const float* __restrict__ als,
                                const float* __restrict__ ald,
                                float* __restrict__ ev, unsigned* __restrict__ mkey) {
    const int tid = blockIdx.x * blockDim.x + threadIdx.x;
    if (tid >= Etot * HEADS) return;
    const int e  = tid >> 2;
    const int hd = tid & 3;
    long long s, d;
    if (e < E) { s = ei[e]; d = ei[E + e]; }
    else       { s = d = (long long)(e - E); }   // self-loop
    float v = als[(size_t)s * HEADS + hd] + ald[(size_t)d * HEADS + hd];
    v = (v > 0.f) ? v : 0.2f * v;                // leaky relu, slope 0.2
    ev[tid] = v;
    atomicMax(&mkey[(size_t)d * HEADS + hd], f2key(v));
}

// ---------------------------------------------------------------------------
// edge pass 2: ex = exp(e - m[dst]); denom[dst,h] += ex;
// num[dst,h,c] += ex * h[src,h,c].  One wave per edge; lane l covers channels
// l, l+32, l+64, l+96 (one per head).
// ---------------------------------------------------------------------------
__global__ __launch_bounds__(256)
void edge_accum_kernel(const long long* __restrict__ ei, int E, int Etot,
                       const float* __restrict__ ev,
                       const unsigned* __restrict__ mkey,
                       const float* __restrict__ h,
                       float* __restrict__ denom, float* __restrict__ num) {
    const int gw   = (blockIdx.x * blockDim.x + threadIdx.x) >> 5;
    const int lane = threadIdx.x & 31;
    if (gw >= Etot) return;
    long long s, d;
    if (gw < E) { s = ei[gw]; d = ei[E + gw]; }
    else        { s = d = (long long)(gw - E); }

    float ex[HEADS];
    #pragma unroll
    for (int hd = 0; hd < HEADS; ++hd) {
        const float m = key2f(mkey[(size_t)d * HEADS + hd]);
        ex[hd] = __expf(ev[(size_t)gw * HEADS + hd] - m);
    }
    if (lane < HEADS)
        atomicAdd(&denom[(size_t)d * HEADS + lane], ex[lane]);
    #pragma unroll
    for (int hd = 0; hd < HEADS; ++hd) {
        const int c = hd * CH + lane;
        atomicAdd(&num[(size_t)d * HID + c], ex[hd] * h[(size_t)s * HID + c]);
    }
}

// ---------------------------------------------------------------------------
// x[n,c] = relu(num[n,c]/denom[n,c/32] + bg[c])
// ---------------------------------------------------------------------------
__global__ void finalize_kernel(const float* __restrict__ num,
                                const float* __restrict__ denom,
                                const float* __restrict__ bg,
                                float* __restrict__ x, int Nn) {
    const int tid = blockIdx.x * blockDim.x + threadIdx.x;
    if (tid >= Nn * HID) return;
    const int n = tid >> 7;
    const int c = tid & 127;
    float v = num[tid] / denom[(size_t)n * HEADS + (c >> 5)] + bg[c];
    x[tid] = (v > 0.f) ? v : 0.f;
}

// ---------------------------------------------------------------------------
// global add pool: out[batch[n], c] += x[n, c]
// ---------------------------------------------------------------------------
__global__ void pool_kernel(const float* __restrict__ x,
                            const long long* __restrict__ batch,
                            float* __restrict__ out, int Nn) {
    const int tid = blockIdx.x * blockDim.x + threadIdx.x;
    if (tid >= Nn * HID) return;
    const int n = tid >> 7;
    const int c = tid & 127;
    atomicAdd(&out[(size_t)batch[n] * HID + c], x[tid]);
}

// ---------------------------------------------------------------------------

extern "C" void kernel_launch(void* const* d_in, const int* in_sizes, int n_in,
                              void* d_out, int out_size, void* d_ws, size_t ws_size,
                              hipStream_t stream) {
    const float*     nf    = (const float*)d_in[0];      // [N,1024]
    const long long* ei    = (const long long*)d_in[1];  // [2,E] int64
    const long long* batch = (const long long*)d_in[2];  // [N] int64
    const float*     W1    = (const float*)d_in[3];      // [1024,128]
    const float*     b1    = (const float*)d_in[4];      // [128]
    const float*     Wg    = (const float*)d_in[5];      // [2,128,128]
    const float*     a_src = (const float*)d_in[6];      // [2,4,32]
    const float*     a_dst = (const float*)d_in[7];      // [2,4,32]
    const float*     bg    = (const float*)d_in[8];      // [2,128]

    const int N    = in_sizes[2];
    const int E    = in_sizes[1] / 2;
    const int Etot = E + N;
    const int IN_DIM = in_sizes[0] / N;                  // 1024

    // workspace layout (floats)
    float* ws    = (float*)d_ws;
    float* x     = ws;                                   // [N,128]
    float* h     = x   + (size_t)N * HID;                // [N,128]
    float* num   = h   + (size_t)N * HID;                // [N,128]
    float* als   = num + (size_t)N * HID;                // [N,4]
    float* ald   = als + (size_t)N * HEADS;              // [N,4]
    float* denom = ald + (size_t)N * HEADS;              // [N,4]
    unsigned* mkey = (unsigned*)(denom + (size_t)N * HEADS); // [N,4]
    float* ev    = (float*)(mkey + (size_t)N * HEADS);   // [Etot,4]

    const int TPB = 256;
    const int gemmBlocks = (N + TM - 1) / TM;

    // layer 0 featurizer: x = gelu(nf @ W1 + b1)
    gemm_wmma_bf16<1><<<gemmBlocks, TPB, 0, stream>>>(nf, W1, b1, x, N, IN_DIM);

    for (int l = 0; l < 2; ++l) {
        const float* Wl = Wg + (size_t)l * HID * HID;
        // h = x @ Wg[l]
        gemm_wmma_bf16<0><<<gemmBlocks, TPB, 0, stream>>>(x, Wl, nullptr, h, N, HID);
        // per-node attention logits
        alpha_kernel<<<(N * HEADS + TPB - 1) / TPB, TPB, 0, stream>>>(
            h, a_src + (size_t)l * HEADS * CH, a_dst + (size_t)l * HEADS * CH,
            als, ald, N);
        // init reductions
        fill_u32_kernel<<<(N * HEADS + TPB - 1) / TPB, TPB, 0, stream>>>(
            mkey, KEY_NEG_INF, N * HEADS);
        fill_f32_kernel<<<(N * HEADS + TPB - 1) / TPB, TPB, 0, stream>>>(
            denom, 0.f, N * HEADS);
        fill_f32_kernel<<<(N * HID + TPB - 1) / TPB, TPB, 0, stream>>>(
            num, 0.f, N * HID);
        // segment max
        edge_max_kernel<<<(Etot * HEADS + TPB - 1) / TPB, TPB, 0, stream>>>(
            ei, E, Etot, als, ald, ev, mkey);
        // exp / weighted aggregate (one wave per edge)
        {
            const long long threads = (long long)Etot * 32;
            edge_accum_kernel<<<(unsigned)((threads + TPB - 1) / TPB), TPB, 0, stream>>>(
                ei, E, Etot, ev, mkey, h, denom, num);
        }
        // x = relu(num/denom + bg[l])
        finalize_kernel<<<(N * HID + TPB - 1) / TPB, TPB, 0, stream>>>(
            num, denom, bg + (size_t)l * HID, x, N);
    }

    // global add pool
    fill_f32_kernel<<<(out_size + TPB - 1) / TPB, TPB, 0, stream>>>(
        (float*)d_out, 0.f, out_size);
    pool_kernel<<<(N * HID + TPB - 1) / TPB, TPB, 0, stream>>>(
        x, batch, (float*)d_out, N);
}
